// TrashNet_6485400616961
// MI455X (gfx1250) — compile-verified, matrix operand order checked by reading, not run on hardware
//
#include <hip/hip_runtime.h>
#include <hip/hip_bf16.h>

// ---------------------------------------------------------------------------
// GatedGraphConv x2 + GRU + edge dot for MI455X (gfx1250, wave32).
// Dense GEMMs use v_wmma_f32_16x16x32_bf16 (f32 accumulate), D padded 30->32
// so K is exactly one WMMA step. Edge phase is L2-resident gather/scatter.
// ---------------------------------------------------------------------------

#define NN 100000
#define EE 3200000
#define DD 30
#define KK 3
#define SP 32   // padded feature stride (one 128B line per node row)

typedef __attribute__((ext_vector_type(16))) __bf16 v16bf;
typedef __attribute__((ext_vector_type(8)))  float  v8f;

union BF16x16 { v16bf v; unsigned short u[16]; };

__device__ __forceinline__ unsigned short f2bf(float f) {
  unsigned int u = __float_as_uint(f);
  unsigned int r = u + 0x7FFFu + ((u >> 16) & 1u);   // round-to-nearest-even
  return (unsigned short)(r >> 16);
}

__device__ __forceinline__ float sigmoidf_(float x) {
  return 1.0f / (1.0f + __expf(-x));
}

// A fragment (bf16 16x32) from f32 row-major, row stride SP.
// ISA layout: lane l holds row M=l&15; lanes<16: K={0..7,16..23}, lanes>=16:
// K={8..15,24..31}; element e<8 -> first K-octet, e>=8 -> second.
__device__ __forceinline__ v16bf load_a_frag(const float* __restrict__ base,
                                             int row0, int lane) {
  int m = lane & 15, hi = lane >> 4;
  const float* rp = base + (size_t)(row0 + m) * SP;
  BF16x16 A;
#pragma unroll
  for (int e = 0; e < 8; ++e) A.u[e]     = f2bf(rp[hi * 8 + e]);
#pragma unroll
  for (int e = 0; e < 8; ++e) A.u[8 + e] = f2bf(rp[16 + hi * 8 + e]);
  return A.v;
}

// B fragment (bf16 32x16): B[d][f] = W[(rowoff + f)*DD + d], zero-padded for
// d>=30 or f>=30. Lane l holds column f = col0 + (l&15); K split as A's.
__device__ __forceinline__ v16bf load_b_frag(const float* __restrict__ W,
                                             int rowoff, int col0, int lane) {
  int n = lane & 15, hi = lane >> 4;
  int f = col0 + n;
  bool fv = (f < DD);
  const float* wr = W + (size_t)(rowoff + (fv ? f : 0)) * DD;
  BF16x16 B;
#pragma unroll
  for (int e = 0; e < 16; ++e) {
    int d = (e < 8) ? (hi * 8 + e) : (16 + hi * 8 + (e - 8));
    float val = (fv && d < DD) ? wr[d] : 0.0f;
    B.u[e] = f2bf(val);
  }
  return B.v;
}

// --------------------------- elementwise kernels ---------------------------

__global__ void k_pack(const float* __restrict__ x, float* __restrict__ h) {
  int tid = blockIdx.x * blockDim.x + threadIdx.x;
  if (tid >= NN * SP) return;
  int nrow = tid >> 5;
  int c = tid & 31;
  h[tid] = (c < DD) ? x[(size_t)nrow * DD + c] : 0.0f;
}

__global__ void k_zero(float* __restrict__ p, int count) {
  int tid = blockIdx.x * blockDim.x + threadIdx.x;
  if (tid < count) p[tid] = 0.0f;
}

// --------------------- per-etype linear: t[k] = h @ W[k]^T + b[k] ----------

__global__ void k_etype_linear(const float* __restrict__ h,
                               const float* __restrict__ W,   // [K,D,D]
                               const float* __restrict__ b,   // [K,D]
                               float* __restrict__ t) {       // [K,N,SP]
  int row0 = blockIdx.x * 16;
  int k    = blockIdx.y;
  int lane = threadIdx.x;

  v16bf A  = load_a_frag(h, row0, lane);
  const float* Wk = W + (size_t)k * DD * DD;
  v16bf B0 = load_b_frag(Wk, 0, 0, lane);
  v16bf B1 = load_b_frag(Wk, 0, 16, lane);

  v8f c0 = {}; v8f c1 = {};
  c0 = __builtin_amdgcn_wmma_f32_16x16x32_bf16(false, A, false, B0, (short)0, c0, false, false);
  c1 = __builtin_amdgcn_wmma_f32_16x16x32_bf16(false, A, false, B1, (short)0, c1, false, false);

  int n = lane & 15, hi = lane >> 4;
  float* tb = t + ((size_t)k * NN + row0) * SP;
  const float* bb = b + (size_t)k * DD;
  float blo = bb[n];
  float bhi = (16 + n < DD) ? bb[16 + n] : 0.0f;
#pragma unroll
  for (int r = 0; r < 8; ++r) {
    int M = r + hi * 8;
    tb[(size_t)M * SP + n]      = c0[r] + blo;
    int c = 16 + n;
    tb[(size_t)M * SP + c]      = (c < DD) ? (c1[r] + bhi) : 0.0f;
  }
}

// --------------- edge aggregate: a[dst] += t[etype, src] -------------------

__global__ void k_aggregate(const float* __restrict__ t,
                            const int* __restrict__ src,
                            const int* __restrict__ dst,
                            const int* __restrict__ ety,
                            float* __restrict__ a) {
  int tid = blockIdx.x * blockDim.x + threadIdx.x;
  int e = tid >> 5;
  int f = tid & 31;
  if (e >= EE) return;
  if (f == 0 && e + 2048 < EE) {
    __builtin_prefetch(src + e + 2048, 0, 1);   // global_prefetch_b8
    __builtin_prefetch(dst + e + 2048, 0, 1);
    __builtin_prefetch(ety + e + 2048, 0, 1);
  }
  int s = src[e], d = dst[e], k = ety[e];
  if (f < DD) {
    float m = t[((size_t)k * NN + s) * SP + f];
    atomicAdd(a + (size_t)d * SP + f, m);       // global_atomic_add_f32 (L2-resident)
  }
}

// ------------------------------- GRU cell ----------------------------------

__device__ __forceinline__ void gru_gate(const v16bf& Aa, const v16bf& Ah,
                                         const float* __restrict__ wi,
                                         const float* __restrict__ wh,
                                         const float* __restrict__ bi,
                                         const float* __restrict__ bh,
                                         int g, int lane,
                                         v8f& gi0, v8f& gi1, v8f& gh0, v8f& gh1) {
  v16bf Bi0 = load_b_frag(wi, g * DD, 0, lane);
  v16bf Bi1 = load_b_frag(wi, g * DD, 16, lane);
  v16bf Bh0 = load_b_frag(wh, g * DD, 0, lane);
  v16bf Bh1 = load_b_frag(wh, g * DD, 16, lane);
  v8f z = {};
  gi0 = __builtin_amdgcn_wmma_f32_16x16x32_bf16(false, Aa, false, Bi0, (short)0, z, false, false);
  gi1 = __builtin_amdgcn_wmma_f32_16x16x32_bf16(false, Aa, false, Bi1, (short)0, z, false, false);
  gh0 = __builtin_amdgcn_wmma_f32_16x16x32_bf16(false, Ah, false, Bh0, (short)0, z, false, false);
  gh1 = __builtin_amdgcn_wmma_f32_16x16x32_bf16(false, Ah, false, Bh1, (short)0, z, false, false);
  int n = lane & 15;
  float bil = bi[g * DD + n];
  float bhl = bh[g * DD + n];
  float bih = (16 + n < DD) ? bi[g * DD + 16 + n] : 0.0f;
  float bhh = (16 + n < DD) ? bh[g * DD + 16 + n] : 0.0f;
#pragma unroll
  for (int r = 0; r < 8; ++r) {
    gi0[r] += bil; gi1[r] += bih;
    gh0[r] += bhl; gh1[r] += bhh;
  }
}

__global__ void k_gru(const float* __restrict__ a,
                      const float* __restrict__ h,
                      const float* __restrict__ wi,   // [3D, D]
                      const float* __restrict__ wh,   // [3D, D]
                      const float* __restrict__ bi,   // [3D]
                      const float* __restrict__ bh,   // [3D]
                      float* __restrict__ out,        // [N, SP]
                      int do_relu) {
  int row0 = blockIdx.x * 16;
  int lane = threadIdx.x;
  int n = lane & 15, hi = lane >> 4;

  v16bf Aa = load_a_frag(a, row0, lane);
  v16bf Ah = load_a_frag(h, row0, lane);

  v8f i0, i1, g0, g1;
  // gate r
  gru_gate(Aa, Ah, wi, wh, bi, bh, 0, lane, i0, i1, g0, g1);
  v8f rr0, rr1;
#pragma unroll
  for (int r = 0; r < 8; ++r) { rr0[r] = sigmoidf_(i0[r] + g0[r]); rr1[r] = sigmoidf_(i1[r] + g1[r]); }
  // gate z
  gru_gate(Aa, Ah, wi, wh, bi, bh, 1, lane, i0, i1, g0, g1);
  v8f zz0, zz1;
#pragma unroll
  for (int r = 0; r < 8; ++r) { zz0[r] = sigmoidf_(i0[r] + g0[r]); zz1[r] = sigmoidf_(i1[r] + g1[r]); }
  // gate n + combine
  gru_gate(Aa, Ah, wi, wh, bi, bh, 2, lane, i0, i1, g0, g1);

  float* ob = out + (size_t)row0 * SP;
  const float* hb = h + (size_t)row0 * SP;
#pragma unroll
  for (int r = 0; r < 8; ++r) {
    int M = r + hi * 8;
    {
      float nv = tanhf(i0[r] + rr0[r] * g0[r]);
      float hv = hb[(size_t)M * SP + n];
      float o  = (1.0f - zz0[r]) * nv + zz0[r] * hv;
      if (do_relu) o = fmaxf(o, 0.0f);
      ob[(size_t)M * SP + n] = o;
    }
    {
      int c = 16 + n;
      float o = 0.0f;
      if (c < DD) {
        float nv = tanhf(i1[r] + rr1[r] * g1[r]);
        float hv = hb[(size_t)M * SP + c];
        o = (1.0f - zz1[r]) * nv + zz1[r] * hv;
        if (do_relu) o = fmaxf(o, 0.0f);
      }
      ob[(size_t)M * SP + c] = o;
    }
  }
}

// ------------------------- edge dot (pos + neg fused) ----------------------

__global__ void k_edgedot(const float* __restrict__ h,
                          const int* __restrict__ s1, const int* __restrict__ d1,
                          const int* __restrict__ s2, const int* __restrict__ d2,
                          float* __restrict__ out) {          // [2E]
  int tid = blockIdx.x * blockDim.x + threadIdx.x;
  int e = tid >> 5;
  int f = tid & 31;
  if (e >= 2 * EE) return;
  bool pos = (e < EE);
  int ei = pos ? e : e - EE;
  int s = pos ? s1[ei] : s2[ei];
  int d = pos ? d1[ei] : d2[ei];
  // padded cols 30,31 are zero -> full-32-lane product is safe and coalesced
  float v = h[(size_t)s * SP + f] * h[(size_t)d * SP + f];
  v += __shfl_xor(v, 16, 32);
  v += __shfl_xor(v, 8, 32);
  v += __shfl_xor(v, 4, 32);
  v += __shfl_xor(v, 2, 32);
  v += __shfl_xor(v, 1, 32);
  if (f == 0) out[e] = v;
}

// ------------------------------- launcher ----------------------------------

extern "C" void kernel_launch(void* const* d_in, const int* in_sizes, int n_in,
                              void* d_out, int out_size, void* d_ws, size_t ws_size,
                              hipStream_t stream) {
  const float* x    = (const float*)d_in[0];
  const int*   src  = (const int*)d_in[1];
  const int*   dst  = (const int*)d_in[2];
  const int*   ety  = (const int*)d_in[3];
  const int*   nsrc = (const int*)d_in[4];
  const int*   ndst = (const int*)d_in[5];
  const float* W1   = (const float*)d_in[6];
  const float* b1   = (const float*)d_in[7];
  const float* wi1  = (const float*)d_in[8];
  const float* wh1  = (const float*)d_in[9];
  const float* bi1  = (const float*)d_in[10];
  const float* bh1  = (const float*)d_in[11];
  const float* W2   = (const float*)d_in[12];
  const float* b2   = (const float*)d_in[13];
  const float* wi2  = (const float*)d_in[14];
  const float* wh2  = (const float*)d_in[15];
  const float* bi2  = (const float*)d_in[16];
  const float* bh2  = (const float*)d_in[17];
  float* out = (float*)d_out;

  // workspace: h0 | t(K) | a | h1   = 6 * N * 32 floats (~77 MB)
  float* ws = (float*)d_ws;
  size_t npad = (size_t)NN * SP;
  float* h0 = ws;                 // layer input / final output (reused as h2)
  float* t  = ws + npad;          // [K, N, SP]
  float* a  = ws + npad * 4;      // aggregation buffer
  float* h1 = ws + npad * 5;      // layer-1 output

  dim3 b256(256), b32(32);
  int elem_blocks = (NN * SP) / 256;        // 12500
  int agg_blocks  = (EE * 32) / 256;        // 400000
  int dot_blocks  = (2 * EE * 32) / 256;    // 800000
  int tile_blocks = NN / 16;                // 6250 (exact)

  k_pack<<<elem_blocks, b256, 0, stream>>>(x, h0);

  // ---- layer 1 ----
  k_etype_linear<<<dim3(tile_blocks, KK), b32, 0, stream>>>(h0, W1, b1, t);
  k_zero<<<elem_blocks, b256, 0, stream>>>(a, NN * SP);
  k_aggregate<<<agg_blocks, b256, 0, stream>>>(t, src, dst, ety, a);
  k_gru<<<tile_blocks, b32, 0, stream>>>(a, h0, wi1, wh1, bi1, bh1, h1, 1);

  // ---- layer 2 ----
  k_etype_linear<<<dim3(tile_blocks, KK), b32, 0, stream>>>(h1, W2, b2, t);
  k_zero<<<elem_blocks, b256, 0, stream>>>(a, NN * SP);
  k_aggregate<<<agg_blocks, b256, 0, stream>>>(t, src, dst, ety, a);
  k_gru<<<tile_blocks, b32, 0, stream>>>(a, h1, wi2, wh2, bi2, bh2, h0, 0);

  // ---- edge scores (pos then neg, concatenated) ----
  k_edgedot<<<dot_blocks, b256, 0, stream>>>(h0, src, dst, nsrc, ndst, out);
}
